// TransformerBlock_44959717654816
// MI455X (gfx1250) — compile-verified, matrix operand order checked by reading, not run on hardware
//
#include <hip/hip_runtime.h>

// ---------------------------------------------------------------------------
// Transformer block (pre-LN): local windowed attention + cross attention + FFN
// B=4, S=4096, D=1024, H=16, HD=64, W=128, DFF=4096, context Sk=1024
// Strategy: bf16 WMMA (v_wmma_f32_16x16x32_bf16) for all GEMMs and attention.
// GEMM v2: 128x128 block tile, 32x64 per wave -> 8 WMMAs per K-step.
// ---------------------------------------------------------------------------

#define NH     16
#define HD     64
#define DMODEL 1024
#define DFFN   4096
#define BSZ    4
#define SEQ    4096
#define SKV    1024
#define NT     (BSZ * SEQ)    // 16384 tokens
#define NTC    (BSZ * SKV)    // 4096 context tokens

typedef unsigned short u16;
typedef unsigned int   u32;
typedef __attribute__((ext_vector_type(4)))  u32    u32x4;
typedef __attribute__((ext_vector_type(16))) __bf16 v16bf;
typedef __attribute__((ext_vector_type(8)))  float  v8f;

union Frag { u32x4 q[2]; v16bf v; };

__device__ __forceinline__ u16 f2bf(float f) {
  u32 u = __float_as_uint(f);
  u32 r = u + 0x7FFFu + ((u >> 16) & 1u);   // round-to-nearest-even
  return (u16)(r >> 16);
}

__device__ __forceinline__ float gelu_tanh(float x) {
  const float k0 = 0.7978845608028654f;   // sqrt(2/pi)
  const float k1 = 0.044715f;
  float inner = k0 * (x + k1 * x * x * x);
  return 0.5f * x * (1.0f + tanhf(inner));
}

// ----------------------------- fp32 -> bf16 --------------------------------
__global__ void cvt_f32_bf16(const float* __restrict__ in, u16* __restrict__ out,
                             long long n) {
  long long i = (long long)blockIdx.x * blockDim.x + threadIdx.x;
  if (i < n) out[i] = f2bf(in[i]);
}

// ------------------------------ LayerNorm ----------------------------------
// one token (row of `d` floats) per 256-thread block; bf16 output
__global__ __launch_bounds__(256) void layernorm_bf16(
    const float* __restrict__ x, const float* __restrict__ g,
    const float* __restrict__ b, u16* __restrict__ out, int d) {
  const int tok = blockIdx.x;
  const float* row = x + (size_t)tok * d;
  float vals[4];
  float s = 0.f, sq = 0.f;
  for (int i = 0; i < 4; i++) {
    float v = row[threadIdx.x + 256 * i];
    vals[i] = v; s += v; sq += v * v;
  }
  __shared__ float rs[256], rq[256];
  rs[threadIdx.x] = s; rq[threadIdx.x] = sq;
  __syncthreads();
  for (int off = 128; off > 0; off >>= 1) {
    if ((int)threadIdx.x < off) {
      rs[threadIdx.x] += rs[threadIdx.x + off];
      rq[threadIdx.x] += rq[threadIdx.x + off];
    }
    __syncthreads();
  }
  float mean = rs[0] / (float)d;
  float var  = rq[0] / (float)d - mean * mean;
  float inv  = rsqrtf(var + 1e-5f);
  u16* orow = out + (size_t)tok * d;
  for (int i = 0; i < 4; i++) {
    int c = threadIdx.x + 256 * i;
    orow[c] = f2bf((vals[i] - mean) * inv * g[c] + b[c]);
  }
}

// ------------------------- V transpose per head ----------------------------
// v:[B][sk][D] (bf16) -> vt:[B][H][HD][sk] (bf16)
__global__ void transpose_v(const u16* __restrict__ v, u16* __restrict__ vt, int sk) {
  long long idx = (long long)blockIdx.x * blockDim.x + threadIdx.x;
  long long total = (long long)BSZ * sk * DMODEL;
  if (idx >= total) return;
  int d = (int)(idx % DMODEL);
  long long t = idx / DMODEL;
  int s = (int)(t % sk);
  int b = (int)(t / sk);
  int h = d / HD, hd = d % HD;
  vt[(((long long)b * NH + h) * HD + hd) * sk + s] = v[idx];
}

// ------------------------------- GEMM --------------------------------------
// C[M,N] = epilogue( A[M,K] (bf16,row-major) @ W[N,K]^T (bf16,row-major) )
// BM=128 BN=128 BK=32; 8 waves (4 M x 2 N), each computing a 32x64 tile
// (2 M-tiles x 4 N-tiles = 8 WMMAs per K-step, A-frags reused 4x).
#define BM 128
#define BN 128
#define BK 32
#define LDT 40   // padded LDS row stride (u16)

__global__ __launch_bounds__(256) void gemm_bf16_wmma(
    const u16* __restrict__ A, const u16* __restrict__ W,
    int M, int N, int K,
    const float* __restrict__ bias, const float* __restrict__ resid,
    int act, int outBf16, float* __restrict__ Cf, u16* __restrict__ Cb) {
  __shared__ u16 At[BM * LDT];
  __shared__ u16 Bt[BN * LDT];
  const int tid  = threadIdx.x;
  const int lane = tid & 31;
  const int wave = tid >> 5;
  const int r    = lane & 15;
  const int half = lane >> 4;
  const int wm   = wave >> 1;   // 0..3 -> M offset wm*32
  const int wn   = wave & 1;    // 0..1 -> N offset wn*64
  const long long mBase = (long long)blockIdx.y * BM;
  const long long nBase = (long long)blockIdx.x * BN;

  v8f acc[2][4];
  for (int a = 0; a < 2; a++)
    for (int bb = 0; bb < 4; bb++)
      for (int i = 0; i < 8; i++) acc[a][bb][i] = 0.f;

  // staging: 128x32 tiles; each thread moves 16 elems of A and 16 of B
  const int srow = tid >> 1;          // 0..127
  const int scol = (tid & 1) * 16;    // 0 | 16

  const u16* aPtr = A + (mBase + srow) * (long long)K + scol;
  const u16* wPtr = W + (nBase + srow) * (long long)K + scol;

  for (int k0 = 0; k0 < K; k0 += BK) {
    const u32x4* asrc = (const u32x4*)(aPtr + k0);
    *(u32x4*)(At + srow * LDT + scol)     = asrc[0];
    *(u32x4*)(At + srow * LDT + scol + 8) = asrc[1];
    const u32x4* bsrc = (const u32x4*)(wPtr + k0);
    *(u32x4*)(Bt + srow * LDT + scol)     = bsrc[0];
    *(u32x4*)(Bt + srow * LDT + scol + 8) = bsrc[1];
    if (k0 + BK < K) {                    // prefetch next K-panel into caches
      __builtin_prefetch(aPtr + k0 + BK, 0, 1);
      __builtin_prefetch(wPtr + k0 + BK, 0, 1);
    }
    __syncthreads();

    Frag af[2], bfr[4];
    for (int t = 0; t < 2; t++) {
      const u16* pa = At + (wm * 32 + t * 16 + r) * LDT + half * 8;
      af[t].q[0] = *(const u32x4*)pa;
      af[t].q[1] = *(const u32x4*)(pa + 16);
    }
    for (int t = 0; t < 4; t++) {
      const u16* pb = Bt + (wn * 64 + t * 16 + r) * LDT + half * 8;
      bfr[t].q[0] = *(const u32x4*)pb;
      bfr[t].q[1] = *(const u32x4*)(pb + 16);
    }
    for (int a = 0; a < 2; a++)
      for (int bb = 0; bb < 4; bb++)
        acc[a][bb] = __builtin_amdgcn_wmma_f32_16x16x32_bf16(
            false, af[a].v, false, bfr[bb].v, (short)0, acc[a][bb], false, false);
    __syncthreads();
  }

  for (int a = 0; a < 2; a++) {
    for (int bb = 0; bb < 4; bb++) {
      for (int i = 0; i < 8; i++) {
        long long row = mBase + wm * 32 + a * 16 + half * 8 + i;
        long long col = nBase + wn * 64 + bb * 16 + r;
        float v = acc[a][bb][i];
        if (bias)  v += bias[col];
        if (act)   v = gelu_tanh(v);
        if (resid) v += resid[row * N + col];
        if (outBf16) Cb[row * N + col] = f2bf(v);
        else         Cf[row * N + col] = v;
      }
    }
  }
}

// --------------------------- Flash attention -------------------------------
// Per block: (b, h, 128-query tile). Per wave: 16-query strip, online softmax
// over key chunks of 128. Q,K: [B][s][D] bf16 (head slice); Vt: [B][H][HD][sk].
__global__ __launch_bounds__(256) void flash_attn_wmma(
    const u16* __restrict__ Q, const u16* __restrict__ K,
    const u16* __restrict__ Vt, u16* __restrict__ O,
    int sqTot, int skTot, int kLen, int isLocal) {
  __shared__ u16 Plds[8][16][128];
  const int tid  = threadIdx.x;
  const int lane = tid & 31;
  const int w    = tid >> 5;
  const int r    = lane & 15;
  const int half = lane >> 4;

  const int nQT = sqTot / 128;
  const int bx  = blockIdx.x;
  const int qt  = bx % nQT;
  const int h   = (bx / nQT) % NH;
  const int b   = bx / (nQT * NH);

  const int qRow0 = qt * 128 + w * 16;
  const int kb0   = isLocal ? qt * 128 : 0;

  float m8[8], l8[8];
  for (int i = 0; i < 8; i++) { m8[i] = -1e30f; l8[i] = 0.f; }
  v8f oacc[4];
  for (int t = 0; t < 4; t++)
    for (int i = 0; i < 8; i++) oacc[t][i] = 0.f;

  const u16* Qlane = Q + ((size_t)b * sqTot + qRow0 + r) * DMODEL + h * HD;

  const int nChunks = kLen / 128;
  for (int c = 0; c < nChunks; c++) {
    const int keyBase = kb0 + c * 128;

    // ---- scores S = (Q K^T) * 1/sqrt(HD), 16x128 per wave ----
    v8f sacc[8];
    for (int nt = 0; nt < 8; nt++)
      for (int i = 0; i < 8; i++) sacc[nt][i] = 0.f;
    for (int ks = 0; ks < 2; ks++) {        // HD=64 -> 2 K-steps of 32
      Frag qf;
      const u16* qp = Qlane + ks * 32 + half * 8;
      qf.q[0] = *(const u32x4*)qp;
      qf.q[1] = *(const u32x4*)(qp + 16);
      for (int nt = 0; nt < 8; nt++) {
        Frag kf;
        const u16* kp = K + ((size_t)b * skTot + keyBase + nt * 16 + r) * DMODEL
                          + h * HD + ks * 32 + half * 8;
        kf.q[0] = *(const u32x4*)kp;
        kf.q[1] = *(const u32x4*)(kp + 16);
        sacc[nt] = __builtin_amdgcn_wmma_f32_16x16x32_bf16(
            false, qf.v, false, kf.v, (short)0, sacc[nt], false, false);
      }
    }

    // ---- online softmax (rows live on 16-lane half-groups) ----
    float cmax[8];
    for (int i = 0; i < 8; i++) cmax[i] = -1e30f;
    for (int nt = 0; nt < 8; nt++)
      for (int i = 0; i < 8; i++) {
        sacc[nt][i] *= 0.125f;              // 1/sqrt(64)
        cmax[i] = fmaxf(cmax[i], sacc[nt][i]);
      }
    for (int s = 1; s < 16; s <<= 1)
      for (int i = 0; i < 8; i++)
        cmax[i] = fmaxf(cmax[i], __shfl_xor(cmax[i], s, 32));

    float alpha[8], mnew[8], csum[8];
    for (int i = 0; i < 8; i++) {
      mnew[i]  = fmaxf(m8[i], cmax[i]);
      alpha[i] = __expf(m8[i] - mnew[i]);
      m8[i]    = mnew[i];
      csum[i]  = 0.f;
    }
    for (int nt = 0; nt < 8; nt++)
      for (int i = 0; i < 8; i++) {
        float p = __expf(sacc[nt][i] - mnew[i]);
        csum[i] += p;
        Plds[w][half * 8 + i][nt * 16 + r] = f2bf(p);
      }
    for (int s = 1; s < 16; s <<= 1)
      for (int i = 0; i < 8; i++)
        csum[i] += __shfl_xor(csum[i], s, 32);
    for (int i = 0; i < 8; i++) l8[i] = l8[i] * alpha[i] + csum[i];
    for (int t = 0; t < 4; t++)
      for (int i = 0; i < 8; i++) oacc[t][i] *= alpha[i];

    // ---- O += P @ V  (K-dim = 128 keys -> 4 steps of 32) ----
    for (int ks2 = 0; ks2 < 4; ks2++) {
      Frag pf;
      const u16* pp = &Plds[w][r][ks2 * 32 + half * 8];
      pf.q[0] = *(const u32x4*)pp;
      pf.q[1] = *(const u32x4*)(pp + 16);
      for (int nt2 = 0; nt2 < 4; nt2++) {
        Frag vf;
        const u16* vp = Vt + (((size_t)b * NH + h) * HD + nt2 * 16 + r) * skTot
                           + keyBase + ks2 * 32 + half * 8;
        vf.q[0] = *(const u32x4*)vp;
        vf.q[1] = *(const u32x4*)(vp + 16);
        oacc[nt2] = __builtin_amdgcn_wmma_f32_16x16x32_bf16(
            false, pf.v, false, vf.v, (short)0, oacc[nt2], false, false);
      }
    }
  }

  for (int i = 0; i < 8; i++) {
    float inv = 1.f / l8[i];
    int row = qRow0 + half * 8 + i;
    for (int t = 0; t < 4; t++) {
      int col = h * HD + t * 16 + r;
      O[((size_t)b * sqTot + row) * DMODEL + col] = f2bf(oacc[t][i] * inv);
    }
  }
}

// ------------------------------- driver ------------------------------------
extern "C" void kernel_launch(void* const* d_in, const int* in_sizes, int n_in,
                              void* d_out, int out_size, void* d_ws, size_t ws_size,
                              hipStream_t stream) {
  const float* x    = (const float*)d_in[0];
  const float* ctx  = (const float*)d_in[1];
  const float* Wq_l = (const float*)d_in[2];
  const float* Wk_l = (const float*)d_in[3];
  const float* Wv_l = (const float*)d_in[4];
  const float* Wo_l = (const float*)d_in[5];
  const float* Wq_c = (const float*)d_in[6];
  const float* Wk_c = (const float*)d_in[7];
  const float* Wv_c = (const float*)d_in[8];
  const float* Wo_c = (const float*)d_in[9];
  const float* W1   = (const float*)d_in[10];
  const float* b1   = (const float*)d_in[11];
  const float* W2   = (const float*)d_in[12];
  const float* b2   = (const float*)d_in[13];
  const float* ln1g = (const float*)d_in[14];
  const float* ln1b = (const float*)d_in[15];
  const float* ln2g = (const float*)d_in[16];
  const float* ln2b = (const float*)d_in[17];
  const float* ln3g = (const float*)d_in[18];
  const float* ln3b = (const float*)d_in[19];
  float* out = (float*)d_out;

  char* wsp = (char*)d_ws;
  size_t off = 0;
  auto alloc16 = [&](size_t elems) -> u16* {
    u16* p = (u16*)(wsp + off);
    off = (off + elems * sizeof(u16) + 255) & ~(size_t)255;
    return p;
  };
  u16* wql = alloc16((size_t)DMODEL * DMODEL);
  u16* wkl = alloc16((size_t)DMODEL * DMODEL);
  u16* wvl = alloc16((size_t)DMODEL * DMODEL);
  u16* wol = alloc16((size_t)DMODEL * DMODEL);
  u16* wqc = alloc16((size_t)DMODEL * DMODEL);
  u16* wkc = alloc16((size_t)DMODEL * DMODEL);
  u16* wvc = alloc16((size_t)DMODEL * DMODEL);
  u16* woc = alloc16((size_t)DMODEL * DMODEL);
  u16* w1b = alloc16((size_t)DFFN * DMODEL);
  u16* w2b = alloc16((size_t)DMODEL * DFFN);
  u16* aln = alloc16((size_t)NT * DMODEL);
  u16* qb  = alloc16((size_t)NT * DMODEL);   // } qb..vt contiguous: reused
  u16* kb  = alloc16((size_t)NT * DMODEL);   // } as 134MB FFN intermediate
  u16* vb  = alloc16((size_t)NT * DMODEL);
  u16* vt  = alloc16((size_t)NT * DMODEL);
  u16* cxb = alloc16((size_t)NT * DMODEL);   // attention context output
  u16* ccb = alloc16((size_t)NTC * DMODEL);  // bf16 context tokens
  u16* midb = qb;                            // [NT, DFFN] bf16 alias

  auto cvt = [&](const float* src, u16* dst, long long n) {
    cvt_f32_bf16<<<(unsigned)((n + 255) / 256), 256, 0, stream>>>(src, dst, n);
  };
  auto gemm = [&](const u16* A, const u16* W, int M, int N, int K,
                  const float* bias, const float* resid, int act,
                  int outBf, float* Cf, u16* Cb) {
    dim3 g((unsigned)(N / BN), (unsigned)(M / BM));
    gemm_bf16_wmma<<<g, 256, 0, stream>>>(A, W, M, N, K, bias, resid, act, outBf, Cf, Cb);
  };

  // weight + context conversion
  const long long DD = (long long)DMODEL * DMODEL;
  cvt(Wq_l, wql, DD); cvt(Wk_l, wkl, DD); cvt(Wv_l, wvl, DD); cvt(Wo_l, wol, DD);
  cvt(Wq_c, wqc, DD); cvt(Wk_c, wkc, DD); cvt(Wv_c, wvc, DD); cvt(Wo_c, woc, DD);
  cvt(W1, w1b, (long long)DFFN * DMODEL);
  cvt(W2, w2b, (long long)DMODEL * DFFN);
  cvt(ctx, ccb, (long long)NTC * DMODEL);

  // ---- 1) local self-attention: h = x + attn(LN1(x)) @ Wo^T ----
  layernorm_bf16<<<NT, 256, 0, stream>>>(x, ln1g, ln1b, aln, DMODEL);
  gemm(aln, wql, NT, DMODEL, DMODEL, nullptr, nullptr, 0, 1, nullptr, qb);
  gemm(aln, wkl, NT, DMODEL, DMODEL, nullptr, nullptr, 0, 1, nullptr, kb);
  gemm(aln, wvl, NT, DMODEL, DMODEL, nullptr, nullptr, 0, 1, nullptr, vb);
  {
    long long n = (long long)NT * DMODEL;
    transpose_v<<<(unsigned)((n + 255) / 256), 256, 0, stream>>>(vb, vt, SEQ);
  }
  flash_attn_wmma<<<BSZ * NH * (SEQ / 128), 256, 0, stream>>>(
      qb, kb, vt, cxb, SEQ, SEQ, 128, 1);
  gemm(cxb, wol, NT, DMODEL, DMODEL, nullptr, x, 0, 0, out, nullptr);  // out = h1

  // ---- 2) cross-attention: h = h + attn(LN2(h), context) @ Wo^T ----
  layernorm_bf16<<<NT, 256, 0, stream>>>(out, ln2g, ln2b, aln, DMODEL);
  gemm(aln, wqc, NT,  DMODEL, DMODEL, nullptr, nullptr, 0, 1, nullptr, qb);
  gemm(ccb, wkc, NTC, DMODEL, DMODEL, nullptr, nullptr, 0, 1, nullptr, kb);
  gemm(ccb, wvc, NTC, DMODEL, DMODEL, nullptr, nullptr, 0, 1, nullptr, vb);
  {
    long long n = (long long)NTC * DMODEL;
    transpose_v<<<(unsigned)((n + 255) / 256), 256, 0, stream>>>(vb, vt, SKV);
  }
  flash_attn_wmma<<<BSZ * NH * (SEQ / 128), 256, 0, stream>>>(
      qb, kb, vt, cxb, SEQ, SKV, SKV, 0);
  gemm(cxb, woc, NT, DMODEL, DMODEL, nullptr, out, 0, 0, out, nullptr); // out = h2

  // ---- 3) FFN: out = h + GELU(LN3(h) @ W1^T + b1) @ W2^T + b2 ----
  layernorm_bf16<<<NT, 256, 0, stream>>>(out, ln3g, ln3b, aln, DMODEL);
  gemm(aln,  w1b, NT, DFFN,   DMODEL, b1, nullptr, 1, 1, nullptr, midb);
  gemm(midb, w2b, NT, DMODEL, DFFN,   b2, out,     0, 0, out, nullptr);
}
